// Cherenkov_GPT_43645457662495
// MI455X (gfx1250) — matrix operand order repaired and checked by reading, not compiled, not run on hardware
//
#include <hip/hip_runtime.h>
#include <hip/hip_bf16.h>

typedef __bf16 bf16;
typedef bf16  v16bf __attribute__((ext_vector_type(16)));
typedef float v8f   __attribute__((ext_vector_type(8)));

#define V_   6147
#define TV_  5923
#define D_   1024
#define S_   512
#define B_   16
#define BS_  8192      /* B*S */
#define DFF_ 2048
#define HU_  128
#define LMLD 6272      /* lm_w padded cols: 49*128 */

#if defined(__has_builtin)
#if __has_builtin(__builtin_amdgcn_global_load_async_to_lds_b128)
#define ASYNC_LDS 1
#endif
#endif

// exact pointee type expected by the async-LDS builtins: int __vector(4)
typedef int vs4i __attribute__((__vector_size__(4 * sizeof(int))));
typedef __attribute__((address_space(1))) vs4i as1_vs4i;
typedef __attribute__((address_space(3))) vs4i as3_vs4i;

__device__ __forceinline__ void wait_async0() {
#if defined(__has_builtin) && __has_builtin(__builtin_amdgcn_s_wait_asynccnt)
  __builtin_amdgcn_s_wait_asynccnt(0);
#else
  asm volatile("s_wait_asynccnt 0x0" ::: "memory");
#endif
}

union Frag { v16bf v; unsigned u[8]; };

// ---- WMMA wave32 fragment index patterns (CDNA5 ISA 7.12.2, 16-bit data) ----
// A matrix 16x32: lane = {half(1),m(4)}; VGPR p holds K pair starting at:
__device__ __forceinline__ int kApat(int p, int half) {
  return (p < 4) ? (2 * p + 8 * half) : (16 + 2 * (p - 4) + 8 * half);
}
// B matrix 32x16: lane = {half(1),n(4)}; VGPR p holds K pair starting at:
__device__ __forceinline__ int kBpat(int p, int half) {
  return (p < 4) ? (2 * p + 8 * half) : (16 + 2 * (p - 4) + 8 * half);
}

__device__ __forceinline__ float gelu_exact(float x) {
  return 0.5f * x * (1.0f + erff(x * 0.70710678118654752f));
}

// ============  fp32 -> bf16 conversion with column padding (zero fill)  ============
__global__ __launch_bounds__(256) void convert_pad_kernel(
    const float* __restrict__ in, bf16* __restrict__ out, int K, int N, int ldb) {
  long total = (long)K * ldb;
  long i = (long)blockIdx.x * blockDim.x + threadIdx.x;
  long stride = (long)gridDim.x * blockDim.x;
  for (; i < total; i += stride) {
    long k = i / ldb;
    int n = (int)(i - k * ldb);
    out[i] = (n < N) ? (bf16)in[k * N + n] : (bf16)0.0f;
  }
}

// =====================  embedding + kinematic features  =====================
__global__ __launch_bounds__(256) void embed_kernel(
    const int* __restrict__ tokens, const int* __restrict__ times,
    const float* __restrict__ kin,
    const float* __restrict__ tokE, const float* __restrict__ posE,
    const float* __restrict__ tposE, const float* __restrict__ timeE,
    const float* __restrict__ momw, const float* __restrict__ momb,
    const float* __restrict__ thw, const float* __restrict__ thb,
    float* __restrict__ x, float* __restrict__ t) {
  int row = blockIdx.x;
  int d0 = threadIdx.x * 4;
  int b = row / S_, s = row - b * S_;
  int tok = tokens[row], tim = times[row];
  float km = kin[b * 2 + 0], kt = kin[b * 2 + 1];
#pragma unroll
  for (int j = 0; j < 4; ++j) {
    int d = d0 + j;
    float xe = tokE[(size_t)tok * D_ + d] + posE[(size_t)s * D_ + d] +
               km * momw[d] + momb[d] + kt * thw[d] + thb[d];
    float te = timeE[(size_t)tim * D_ + d] + tposE[(size_t)s * D_ + d];
    x[(size_t)row * D_ + d] = xe;
    t[(size_t)row * D_ + d] = te;
  }
}

// =====================  LayerNorm (row of 1024) -> bf16  =====================
__global__ __launch_bounds__(256) void ln_bf16_kernel(
    const float* __restrict__ x, const float* __restrict__ g,
    const float* __restrict__ b, bf16* __restrict__ out) {
  __shared__ float r1[256], r2[256];
  int row = blockIdx.x, tid = threadIdx.x;
  float4 v = *(const float4*)(x + (size_t)row * D_ + tid * 4);
  float s = v.x + v.y + v.z + v.w;
  float ss = v.x * v.x + v.y * v.y + v.z * v.z + v.w * v.w;
  r1[tid] = s; r2[tid] = ss;
  __syncthreads();
  for (int o = 128; o > 0; o >>= 1) {
    if (tid < o) { r1[tid] += r1[tid + o]; r2[tid] += r2[tid + o]; }
    __syncthreads();
  }
  float mean = r1[0] * (1.0f / D_);
  float var  = r2[0] * (1.0f / D_) - mean * mean;
  float rstd = rsqrtf(var + 1e-5f);
  int d = tid * 4;
  out[(size_t)row * D_ + d + 0] = (bf16)((v.x - mean) * rstd * g[d + 0] + b[d + 0]);
  out[(size_t)row * D_ + d + 1] = (bf16)((v.y - mean) * rstd * g[d + 1] + b[d + 1]);
  out[(size_t)row * D_ + d + 2] = (bf16)((v.z - mean) * rstd * g[d + 2] + b[d + 2]);
  out[(size_t)row * D_ + d + 3] = (bf16)((v.w - mean) * rstd * g[d + 3] + b[d + 3]);
}

// ========== per-head L2 normalize (+ optional learnable scale) -> bf16 ==========
__global__ __launch_bounds__(256) void l2norm_kernel(
    const float* __restrict__ in, const float* __restrict__ gptr,
    bf16* __restrict__ out, int H, int dh) {
  int gw = blockIdx.x * 8 + (threadIdx.x >> 5);
  int lane = threadIdx.x & 31;
  int row = gw / H, h = gw - row * H;
  const float* p = in + (size_t)row * D_ + h * dh;
  int per = dh >> 5;  // 2 or 4
  float vals[4];
  float ss = 0.f;
  for (int j = 0; j < per; ++j) {
    float v = p[lane + j * 32];
    vals[j] = v;
    ss += v * v;
  }
  for (int o = 16; o > 0; o >>= 1) ss += __shfl_xor(ss, o, 32);
  float g = gptr ? gptr[0] : 1.0f;
  float sc = g / fmaxf(sqrtf(ss), 1e-12f);
  bf16* q = out + (size_t)row * D_ + h * dh;
  for (int j = 0; j < per; ++j) q[lane + j * 32] = (bf16)(vals[j] * sc);
}

// ====== transpose V per head: vt[b,h,c,s] = v[b,s,h*dh+c] (bf16, for PV WMMA) ======
__global__ __launch_bounds__(256) void vtrans_kernel(
    const bf16* __restrict__ v, bf16* __restrict__ vt, int H, int dh) {
  long i = (long)blockIdx.x * 256 + threadIdx.x;  // over BS_*D_
  long row = i >> 10;          // b*S + s
  int c = (int)(i & (D_ - 1)); // 0..1023
  int b = (int)(row >> 9), s = (int)(row & (S_ - 1));
  int h = c / dh, cc = c - h * dh;
  vt[(((size_t)b * H + h) * dh + cc) * S_ + s] = v[i];
}

// =====================  bf16 GEMM with fused epilogue  =====================
// C[M,N] = act(A[M,K] @ B[K,ldb(cols 0..N-1)] + bias + resid)
// act: 0=none 1=gelu 2=relu. B is zero-padded to ldb (multiple of 128).
// Block tile 128x128x32, 8 waves, each wave: 2x4 tiles of v_wmma_f32_16x16x32_bf16.
__global__ __launch_bounds__(256) void gemm_bf16_kernel(
    const bf16* __restrict__ A, const bf16* __restrict__ Bm,
    const float* __restrict__ bias, const float* __restrict__ resid,
    float* __restrict__ outF, bf16* __restrict__ outBF,
    int M, int N, int ldb, int K, int act) {
  __shared__ bf16 As[2][128 * 32];  // [m][k]
  __shared__ bf16 Bs[2][128 * 32];  // transposed: [n][k]
  const int tid = threadIdx.x;
  const int lane = tid & 31, wave = tid >> 5;
  const int half = lane >> 4, l15 = lane & 15;
  const int wr = wave & 3, wc = wave >> 2;
  const int bm = blockIdx.y * 128, bn = blockIdx.x * 128;

  v8f acc[2][4];
#pragma unroll
  for (int mi = 0; mi < 2; ++mi)
#pragma unroll
    for (int ni = 0; ni < 4; ++ni)
#pragma unroll
      for (int e = 0; e < 8; ++e) acc[mi][ni][e] = 0.0f;

  const int arow = tid >> 1;            // A: 2 threads per row, 16 bf16 each
  const int ach = (tid & 1) * 16;
  const int bkp = (tid >> 4) * 2;       // B: K-pair base (0,2,..,30)
  const int bng = (tid & 15) * 8;       // B: 8-column group

  const bf16* aBase = A + (size_t)(bm + arow) * K + ach;
  const bf16* bBase = Bm + (size_t)bkp * ldb + bn + bng;

  auto loadA = [&](int stage, int kt) {
    const bf16* src = aBase + kt * 32;
#if defined(ASYNC_LDS)
    __builtin_amdgcn_global_load_async_to_lds_b128(
        (as1_vs4i*)(void*)src,
        (as3_vs4i*)&As[stage][arow * 32 + ach], 0, 0);
    __builtin_amdgcn_global_load_async_to_lds_b128(
        (as1_vs4i*)(void*)(src + 8),
        (as3_vs4i*)&As[stage][arow * 32 + ach + 8], 0, 0);
#else
    *(uint4*)&As[stage][arow * 32 + ach] = *(const uint4*)(src);
    *(uint4*)&As[stage][arow * 32 + ach + 8] = *(const uint4*)(src + 8);
#endif
  };
  auto loadB = [&](int stage, int kt) {
    const bf16* s0 = bBase + (size_t)kt * 32 * ldb;
    union { uint4 q; unsigned short h[8]; } r0, r1;
    r0.q = *(const uint4*)s0;
    r1.q = *(const uint4*)(s0 + ldb);
#pragma unroll
    for (int i = 0; i < 8; ++i) {
      unsigned d = (unsigned)r0.h[i] | ((unsigned)r1.h[i] << 16);
      *(unsigned*)&Bs[stage][(bng + i) * 32 + bkp] = d;
    }
  };

  int nkt = K >> 5;
  loadA(0, 0);
  loadB(0, 0);
#if defined(ASYNC_LDS)
  wait_async0();
#endif
  __syncthreads();

  for (int kt = 0; kt < nkt; ++kt) {
    int cur = kt & 1, nxt = cur ^ 1;
    if (kt + 1 < nkt) {
      __builtin_prefetch(aBase + (kt + 2) * 32, 0, 1);
      __builtin_prefetch(bBase + (size_t)(kt + 2) * 32 * ldb, 0, 1);
      loadA(nxt, kt + 1);
      loadB(nxt, kt + 1);
    }
    Frag af[2], bfr[4];
#pragma unroll
    for (int mi = 0; mi < 2; ++mi) {
      int row = wr * 32 + mi * 16 + l15;
#pragma unroll
      for (int p = 0; p < 8; ++p)
        af[mi].u[p] = *(const unsigned*)&As[cur][row * 32 + kApat(p, half)];
    }
#pragma unroll
    for (int ni = 0; ni < 4; ++ni) {
      int col = wc * 64 + ni * 16 + l15;
#pragma unroll
      for (int p = 0; p < 8; ++p)
        bfr[ni].u[p] = *(const unsigned*)&Bs[cur][col * 32 + kBpat(p, half)];
    }
#pragma unroll
    for (int mi = 0; mi < 2; ++mi)
#pragma unroll
      for (int ni = 0; ni < 4; ++ni)
        acc[mi][ni] = __builtin_amdgcn_wmma_f32_16x16x32_bf16(
            false, af[mi].v, false, bfr[ni].v, (short)0, acc[mi][ni], false, false);
#if defined(ASYNC_LDS)
    wait_async0();
#endif
    __syncthreads();
  }

  // epilogue
#pragma unroll
  for (int mi = 0; mi < 2; ++mi) {
#pragma unroll
    for (int ni = 0; ni < 4; ++ni) {
      int gcol = bn + wc * 64 + ni * 16 + l15;
      if (gcol >= N) continue;
      float bv = bias ? bias[gcol] : 0.0f;
#pragma unroll
      for (int e = 0; e < 8; ++e) {
        int grow = bm + wr * 32 + mi * 16 + e + 8 * half;
        float v = acc[mi][ni][e] + bv;
        if (resid) v += resid[(size_t)grow * N + gcol];
        if (act == 1) v = gelu_exact(v);
        else if (act == 2) v = fmaxf(v, 0.0f);
        if (outF)  outF[(size_t)grow * N + gcol] = v;
        if (outBF) outBF[(size_t)grow * N + gcol] = (bf16)v;
      }
    }
  }
}

// =====================  flash attention (causal + kpm), bf16 WMMA  =====================
// grid: (S/128, H, B), block 256 (8 waves). Each wave owns 16 query rows,
// streams 32-key chunks with online softmax; QK^T and PV via WMMA bf16.
// vt is the per-head transposed V: vt[b,h,c,s].
__global__ __launch_bounds__(256) void attn_kernel(
    const bf16* __restrict__ qb, const bf16* __restrict__ kb,
    const bf16* __restrict__ vt, const unsigned char* __restrict__ kpm,
    bf16* __restrict__ ob, int H, int dh) {
  __shared__ bf16 pbuf[8][16][32];  // wave-private P redistribution buffer
  const int tid = threadIdx.x, lane = tid & 31, wave = tid >> 5;
  const int half = lane >> 4, l15 = lane & 15;
  const int b = blockIdx.z, h = blockIdx.y;
  const int q0 = blockIdx.x * 128 + wave * 16;
  const size_t hbase = (size_t)b * S_ * D_ + (size_t)h * dh;
  const size_t vtbase = ((size_t)b * H + h) * dh * S_;

  const int nks = dh >> 5;  // K-steps of 32 over dh (2 or 4)
  const int nnt = dh >> 4;  // output tiles over dh (4 or 8)

  // preload q A-fragments
  Frag qf[4];
  {
    const bf16* qrp = qb + hbase + (size_t)(q0 + l15) * D_;
    for (int ks = 0; ks < nks; ++ks)
#pragma unroll
      for (int p = 0; p < 8; ++p)
        qf[ks].u[p] = *(const unsigned*)(qrp + ks * 32 + kApat(p, half));
  }

  v8f oacc[8];
#pragma unroll
  for (int ni = 0; ni < 8; ++ni)
#pragma unroll
    for (int e = 0; e < 8; ++e) oacc[ni][e] = 0.0f;
  float mrow[8], lrow[8];
#pragma unroll
  for (int e = 0; e < 8; ++e) { mrow[e] = -3.0e38f; lrow[e] = 0.0f; }

  const int nkc = (q0 + 15) / 32 + 1;  // causal: only chunks with keys <= qmax
  for (int kc = 0; kc < nkc; ++kc) {
    v8f sacc[2];
#pragma unroll
    for (int t = 0; t < 2; ++t)
#pragma unroll
      for (int e = 0; e < 8; ++e) sacc[t][e] = 0.0f;

    // scores for 32 keys (two 16-key tiles)
#pragma unroll
    for (int t = 0; t < 2; ++t) {
      int key = kc * 32 + t * 16 + l15;
      const bf16* kp = kb + hbase + (size_t)key * D_;
      for (int ks = 0; ks < nks; ++ks) {
        Frag kf;
#pragma unroll
        for (int p = 0; p < 8; ++p)
          kf.u[p] = *(const unsigned*)(kp + ks * 32 + kBpat(p, half));
        sacc[t] = __builtin_amdgcn_wmma_f32_16x16x32_bf16(
            false, qf[ks].v, false, kf.v, (short)0, sacc[t], false, false);
      }
    }
    // masking: causal (key > query) and key padding
#pragma unroll
    for (int t = 0; t < 2; ++t) {
      int key = kc * 32 + t * 16 + l15;
      bool pad = kpm[b * S_ + key] != 0;
#pragma unroll
      for (int e = 0; e < 8; ++e) {
        int qr = q0 + e + 8 * half;
        if (key > qr || pad) sacc[t][e] = -1e30f;
      }
    }
    // online softmax update + write P (bf16) to wave-private LDS
#pragma unroll
    for (int e = 0; e < 8; ++e) {
      float tmax = fmaxf(sacc[0][e], sacc[1][e]);
      for (int o = 8; o > 0; o >>= 1) tmax = fmaxf(tmax, __shfl_xor(tmax, o, 32));
      float mnew = fmaxf(mrow[e], tmax);
      float alpha = __expf(mrow[e] - mnew);
      mrow[e] = mnew;
      float p0 = __expf(sacc[0][e] - mnew);
      float p1 = __expf(sacc[1][e] - mnew);
      float s = p0 + p1;
      for (int o = 8; o > 0; o >>= 1) s += __shfl_xor(s, o, 32);
      lrow[e] = lrow[e] * alpha + s;
      for (int ni = 0; ni < nnt; ++ni) oacc[ni][e] *= alpha;
      int r = e + 8 * half;
      pbuf[wave][r][l15] = (bf16)p0;
      pbuf[wave][r][16 + l15] = (bf16)p1;
    }
    // P as A-fragment
    Frag pf;
#pragma unroll
    for (int p = 0; p < 8; ++p)
      pf.u[p] = *(const unsigned*)&pbuf[wave][l15][kApat(p, half)];
    // V as B-fragments from transposed vt (key pairs contiguous), accumulate O
    for (int ni = 0; ni < nnt; ++ni) {
      Frag vf;
      const bf16* vp = vt + vtbase + (size_t)(ni * 16 + l15) * S_ + kc * 32;
#pragma unroll
      for (int p = 0; p < 8; ++p)
        vf.u[p] = *(const unsigned*)(vp + kBpat(p, half));
      oacc[ni] = __builtin_amdgcn_wmma_f32_16x16x32_bf16(
          false, pf.v, false, vf.v, (short)0, oacc[ni], false, false);
    }
  }
  // normalize + write out
#pragma unroll
  for (int e = 0; e < 8; ++e) {
    float inv = 1.0f / lrow[e];
    int qr = q0 + e + 8 * half;
    for (int ni = 0; ni < nnt; ++ni)
      ob[hbase + (size_t)qr * D_ + ni * 16 + l15] = (bf16)(oacc[ni][e] * inv);
  }
}

// =====================  TimeRegression output: relu(y @ w + b)  =====================
__global__ __launch_bounds__(256) void trout_kernel(
    const float* __restrict__ y, const float* __restrict__ w,
    const float* __restrict__ bptr, float* __restrict__ out) {
  int row = blockIdx.x * 8 + (threadIdx.x >> 5);
  int lane = threadIdx.x & 31;
  float s = 0.f;
#pragma unroll
  for (int j = 0; j < 4; ++j) {
    int i = lane + j * 32;
    s += y[(size_t)row * HU_ + i] * w[i];
  }
  for (int o = 16; o > 0; o >>= 1) s += __shfl_xor(s, o, 32);
  if (lane == 0) out[row] = fmaxf(s + bptr[0], 0.0f);
}

// ============================  host orchestration  ============================
extern "C" void kernel_launch(void* const* d_in, const int* in_sizes, int n_in,
                              void* d_out, int out_size, void* d_ws, size_t ws_size,
                              hipStream_t stream) {
  (void)in_sizes; (void)n_in; (void)out_size; (void)ws_size;
  const int* tokens = (const int*)d_in[0];
  const int* times  = (const int*)d_in[1];
  const float* kin  = (const float*)d_in[2];
  const unsigned char* kpm = (const unsigned char*)d_in[3];
  auto P = [&](int i) { return (const float*)d_in[i]; };

  // params leaf order (jax tree: sorted dict keys, lists in order)
  const int qw_i[3]={18,32,46}, kw_i[3]={11,27,41}, vw_i[3]={19,33,47};
  const int cw_i[3]={5,21,35},  cb_i[3]={4,20,34};
  const int w1_i[3]={8,24,38},  w2_i[3]={9,25,39};
  const int b1_i[3]={6,22,36},  b2_i[3]={7,23,37};
  const int g_i[3]={10,26,40};
  const int ln1g_i[3]={17,29,43}, ln1b_i[3]={16,28,42};
  const int ln2g_i[3]={13,31,45}, ln2b_i[3]={12,30,44};
  const int lntg=15, lntb=14, lmb=48, lmw=49, lnfb=50, lnfg=51;
  const int momb=52, momw=53, pose=54, thb=55, thw=56, timee=57, toke=58, tpose=59;
  const int tr_inb=60, tr_inw=61, tr_outb=62, tr_outw=63;
  const int rb1_i[2]={64,68}, rb2_i[2]={65,69}, rw1_i[2]={66,70}, rw2_i[2]={67,71};

  // workspace carve (bump allocator)
  char* wp = (char*)d_ws;
  auto alloc = [&](size_t bytes) { char* r = wp; wp += (bytes + 255) & ~(size_t)255; return r; };
  float* xf   = (float*)alloc((size_t)BS_ * D_ * 4);
  float* tf   = (float*)alloc((size_t)BS_ * D_ * 4);
  float* qfb  = (float*)alloc((size_t)BS_ * D_ * 4);
  float* kfb  = (float*)alloc((size_t)BS_ * D_ * 4);
  bf16* xnbf  = (bf16*)alloc((size_t)BS_ * D_ * 2);
  bf16* tnbf  = (bf16*)alloc((size_t)BS_ * D_ * 2);
  bf16* hbuf  = (bf16*)alloc((size_t)BS_ * DFF_ * 2);
  bf16* qbf   = (bf16*)alloc((size_t)BS_ * D_ * 2);
  bf16* kbf   = (bf16*)alloc((size_t)BS_ * D_ * 2);
  bf16* vbf   = (bf16*)alloc((size_t)BS_ * D_ * 2);
  bf16* vtbf  = (bf16*)alloc((size_t)BS_ * D_ * 2);
  bf16* obf   = (bf16*)alloc((size_t)BS_ * D_ * 2);
  float* ybuf = (float*)alloc((size_t)BS_ * HU_ * 4);
  bf16* ybf   = (bf16*)alloc((size_t)BS_ * HU_ * 2);
  bf16* tbf   = (bf16*)alloc((size_t)BS_ * HU_ * 2);
  // bf16 weight arena (B matrices, column-padded to multiple of 128)
  bf16* qwb[3], *kwb[3], *vwb[3], *cwb[3], *w1b[3], *w2b[3];
  for (int i = 0; i < 3; ++i) {
    qwb[i] = (bf16*)alloc((size_t)D_ * D_ * 2);
    kwb[i] = (bf16*)alloc((size_t)D_ * D_ * 2);
    vwb[i] = (bf16*)alloc((size_t)D_ * D_ * 2);
    cwb[i] = (bf16*)alloc((size_t)D_ * D_ * 2);
    w1b[i] = (bf16*)alloc((size_t)D_ * DFF_ * 2);
    w2b[i] = (bf16*)alloc((size_t)DFF_ * D_ * 2);
  }
  bf16* lmwb = (bf16*)alloc((size_t)D_ * LMLD * 2);
  bf16* inwb = (bf16*)alloc((size_t)D_ * HU_ * 2);
  bf16* rw1b[2], *rw2b[2];
  for (int r = 0; r < 2; ++r) {
    rw1b[r] = (bf16*)alloc((size_t)HU_ * HU_ * 2);
    rw2b[r] = (bf16*)alloc((size_t)HU_ * HU_ * 2);
  }

  auto convpad = [&](int idx, bf16* dst, int K, int N, int ldbp) {
    long total = (long)K * ldbp;
    long g = (total + 255) / 256; if (g > 4096) g = 4096;
    convert_pad_kernel<<<dim3((unsigned)g), 256, 0, stream>>>(P(idx), dst, K, N, ldbp);
  };
  for (int i = 0; i < 3; ++i) {
    convpad(qw_i[i], qwb[i], D_, D_, D_);
    convpad(kw_i[i], kwb[i], D_, D_, D_);
    convpad(vw_i[i], vwb[i], D_, D_, D_);
    convpad(cw_i[i], cwb[i], D_, D_, D_);
    convpad(w1_i[i], w1b[i], D_, DFF_, DFF_);
    convpad(w2_i[i], w2b[i], DFF_, D_, D_);
  }
  convpad(lmw, lmwb, D_, V_, LMLD);
  convpad(tr_inw, inwb, D_, HU_, HU_);
  for (int r = 0; r < 2; ++r) {
    convpad(rw1_i[r], rw1b[r], HU_, HU_, HU_);
    convpad(rw2_i[r], rw2b[r], HU_, HU_, HU_);
  }

  auto gemm = [&](const bf16* A, const bf16* Bw, const float* bias, const float* res,
                  float* oF, bf16* oB, int M, int N, int ldbp, int K, int act) {
    dim3 g((N + 127) / 128, M / 128);
    gemm_bf16_kernel<<<g, 256, 0, stream>>>(A, Bw, bias, res, oF, oB, M, N, ldbp, K, act);
  };

  embed_kernel<<<BS_, 256, 0, stream>>>(tokens, times, kin, P(toke), P(pose),
                                        P(tpose), P(timee), P(momw), P(momb),
                                        P(thw), P(thb), xf, tf);

  for (int blk = 0; blk < 3; ++blk) {
    int H = (blk == 1) ? 16 : 8;
    int dh = D_ / H;
    ln_bf16_kernel<<<BS_, 256, 0, stream>>>(xf, P(ln1g_i[blk]), P(ln1b_i[blk]), xnbf);
    const bf16* qin = xnbf;
    if (blk == 0) {
      ln_bf16_kernel<<<BS_, 256, 0, stream>>>(tf, P(lntg), P(lntb), tnbf);
      qin = tnbf;
    }
    gemm(qin,  qwb[blk], nullptr, nullptr, qfb, nullptr, BS_, D_, D_, D_, 0);
    gemm(xnbf, kwb[blk], nullptr, nullptr, kfb, nullptr, BS_, D_, D_, D_, 0);
    gemm(xnbf, vwb[blk], nullptr, nullptr, nullptr, vbf, BS_, D_, D_, D_, 0);
    l2norm_kernel<<<BS_ * H / 8, 256, 0, stream>>>(qfb, P(g_i[blk]), qbf, H, dh);
    l2norm_kernel<<<BS_ * H / 8, 256, 0, stream>>>(kfb, nullptr, kbf, H, dh);
    vtrans_kernel<<<BS_ * D_ / 256, 256, 0, stream>>>(vbf, vtbf, H, dh);
    attn_kernel<<<dim3(S_ / 128, H, B_), 256, 0, stream>>>(qbf, kbf, vtbf, kpm, obf, H, dh);
    gemm(obf, cwb[blk], P(cb_i[blk]), xf, xf, nullptr, BS_, D_, D_, D_, 0);
    ln_bf16_kernel<<<BS_, 256, 0, stream>>>(xf, P(ln2g_i[blk]), P(ln2b_i[blk]), xnbf);
    gemm(xnbf, w1b[blk], P(b1_i[blk]), nullptr, nullptr, hbuf, BS_, DFF_, DFF_, D_, 1);
    gemm(hbuf, w2b[blk], P(b2_i[blk]), xf, xf, nullptr, BS_, D_, D_, DFF_, 0);
  }

  // final LN -> LM head (logits straight to d_out) + TimeRegression head
  ln_bf16_kernel<<<BS_, 256, 0, stream>>>(xf, P(lnfg), P(lnfb), xnbf);
  gemm(xnbf, lmwb, P(lmb), nullptr, (float*)d_out, nullptr, BS_, V_, LMLD, D_, 0);

  gemm(xnbf, inwb, P(tr_inb), nullptr, ybuf, ybf, BS_, HU_, HU_, D_, 0);
  for (int r = 0; r < 2; ++r) {
    gemm(ybf, rw1b[r], P(rb1_i[r]), nullptr, nullptr, tbf, BS_, HU_, HU_, HU_, 2);
    gemm(tbf, rw2b[r], P(rb2_i[r]), ybuf, ybuf, ybf, BS_, HU_, HU_, HU_, 2);
  }
  trout_kernel<<<BS_ / 8, 256, 0, stream>>>(ybuf, P(tr_outw), P(tr_outb),
                                            (float*)d_out + (size_t)BS_ * V_);
}
